// GraphCON_GAT_33320356282950
// MI455X (gfx1250) — matrix operand order repaired and checked by reading, compiled, not run on hardware
//
#include <hip/hip_runtime.h>
#include <hip/hip_bf16.h>

// ---------------- problem constants (match reference) ----------------
#define NHID    128
#define HEADS   4
#define HC      (HEADS * NHID)   // 512
#define NLAYERS 4
#define NCLASS  40
#define NEG_SLOPE 0.2f

typedef __attribute__((ext_vector_type(16))) __bf16 v16bf;
typedef __attribute__((ext_vector_type(8)))  float  v8f;
typedef __attribute__((ext_vector_type(4)))  unsigned int v4u;
typedef __attribute__((ext_vector_type(4)))  float  v4f;

union Frag16 { v4u q[2]; v16bf v; };

// ---------------- helpers ----------------
__device__ __forceinline__ unsigned short f2bf(float f) {
    unsigned int u = __float_as_uint(f);
    u = (u + 0x7FFFu + ((u >> 16) & 1u)) >> 16;   // round-to-nearest-even
    return (unsigned short)u;
}
// order-preserving float -> uint key for atomicMax over signed floats
__device__ __forceinline__ unsigned int fkey(float f) {
    unsigned int u = __float_as_uint(f);
    return (u & 0x80000000u) ? ~u : (u | 0x80000000u);
}
__device__ __forceinline__ float funkey(unsigned int k) {
    unsigned int u = (k & 0x80000000u) ? (k & 0x7FFFFFFFu) : ~k;
    return __uint_as_float(u);
}

// ---------------- kernels ----------------

// X0 -> Xcur, Ycur, Xbf, and layer-0 slices of X_all / Y_all
__global__ void k_init(const float* __restrict__ x, float* Xcur, float* Ycur,
                       unsigned short* Xbf, float* Xall, float* Yall, int N) {
    int t = blockIdx.x * blockDim.x + threadIdx.x;
    if (t >= N * NHID) return;
    float v = x[t];
    Xcur[t] = v; Ycur[t] = v; Xbf[t] = f2bf(v);
    int n = t / NHID, c = t % NHID;
    Xall[(size_t)n * (NLAYERS + 1) * NHID + c] = v;
    Yall[(size_t)n * (NLAYERS + 1) * NHID + c] = v;
}

// Pack W [128 x 512] (K x N) into per-lane WMMA B-fragment order (bf16 bits).
// B tile = 32x16 (K x N): lanes 0-15 hold col n=lane, K = k0+0..7,16..23 (elem i);
//                          lanes 16-31 hold col n=lane-16, K = k0+8..15,24..31.
__global__ void k_prepB(const float* __restrict__ W, unsigned short* Bpack) {
    int t = blockIdx.x * blockDim.x + threadIdx.x;
    if (t >= 32 * 4 * 32 * 16) return;
    int i    = t & 15;
    int lane = (t >> 4) & 31;
    int ks   = (t >> 9) & 3;
    int ct   = t >> 11;
    int n = ct * 16 + (lane & 15);
    int k = ks * 32 + ((i >> 3) << 4) + ((lane >> 4) << 3) + (i & 7);
    Bpack[t] = f2bf(W[k * HC + n]);
}

// zero the per-layer accumulators (vectorized)
__global__ void k_zero(float* outbuf, unsigned int* mbits, float* den, int N) {
    int t = blockIdx.x * blockDim.x + threadIdx.x;
    if (t < (N * HC) / 4) {
        v4f z = {0.f, 0.f, 0.f, 0.f};
        *reinterpret_cast<v4f*>(outbuf + 4 * (size_t)t) = z;
    }
    if (t < N * HEADS) { mbits[t] = 0u; den[t] = 0.0f; }
}

// h[N,512] = Xbf[N,128] @ Wbf[128,512]; one wave per 16x128 output strip.
// All loads/stores are immediate offsets off per-lane base pointers.
__global__ void __launch_bounds__(32)
k_gemm(const unsigned short* __restrict__ Xbf,
       const unsigned short* __restrict__ Bpack,
       float* __restrict__ hbuf, int N) {
    int lane = threadIdx.x;
    int tm = blockIdx.x;          // row tile (16 rows)
    int cg = blockIdx.y;          // group of 8 col tiles (128 cols)

    v8f acc[8];
#pragma unroll
    for (int j = 0; j < 8; ++j)
        acc[j] = (v8f){0.f,0.f,0.f,0.f,0.f,0.f,0.f,0.f};

    int row = tm * 16 + (lane & 15);
    if (row >= N) row = N - 1;
    // A base: row-major bf16 row, +8 elems for the high half-wave
    const unsigned short* abase = Xbf + (size_t)row * NHID + ((lane >> 4) << 3);
    // B base: fragment-packed, per-lane 16 elems; tile (cg*8+j, ks) at
    // element offset (j*4+ks)*512 from this base
    const unsigned short* bbase = Bpack + (size_t)cg * (8 * 4 * 32 * 16) + lane * 16;

#pragma unroll
    for (int ks = 0; ks < 4; ++ks) {
        Frag16 A;
        A.q[0] = *reinterpret_cast<const v4u*>(abase + ks * 32);       // K lo half
        A.q[1] = *reinterpret_cast<const v4u*>(abase + ks * 32 + 16);  // K hi half
#pragma unroll
        for (int j = 0; j < 8; ++j) {
            Frag16 B;
            const unsigned short* bp = bbase + (j * 4 + ks) * 512;
            B.q[0] = *reinterpret_cast<const v4u*>(bp);
            B.q[1] = *reinterpret_cast<const v4u*>(bp + 8);
            acc[j] = __builtin_amdgcn_wmma_f32_16x16x32_bf16(
                false, A.v, false, B.v, (short)0, acc[j], false, false);
        }
    }

    // C layout: VGPR v, lanes 0-15 -> row v, col=lane; lanes 16-31 -> row v+8
    int rbase = tm * 16 + ((lane >> 4) << 3);
    float* obase = hbuf + (size_t)rbase * HC + cg * 128 + (lane & 15);
    bool full = (rbase + 7) < N;
    if (full) {
#pragma unroll
        for (int j = 0; j < 8; ++j)
#pragma unroll
            for (int v = 0; v < 8; ++v)
                obase[(size_t)v * HC + j * 16] = acc[j][v];
    } else {
#pragma unroll
        for (int j = 0; j < 8; ++j)
#pragma unroll
            for (int v = 0; v < 8; ++v)
                if (rbase + v < N) obase[(size_t)v * HC + j * 16] = acc[j][v];
    }
}

// per-node, per-head attention projections (vectorized dot products)
__global__ void k_att(const float* __restrict__ hbuf,
                      const float* __restrict__ att_src,
                      const float* __restrict__ att_dst,
                      float* a_s, float* a_d, int N) {
    int t = blockIdx.x * blockDim.x + threadIdx.x;
    if (t >= N * HEADS) return;
    int n = t >> 2, h0 = t & 3;
    const v4f* hp = reinterpret_cast<const v4f*>(hbuf + (size_t)n * HC + h0 * NHID);
    const v4f* as = reinterpret_cast<const v4f*>(att_src + h0 * NHID);
    const v4f* ad = reinterpret_cast<const v4f*>(att_dst + h0 * NHID);
    float sa = 0.f, sd = 0.f;
#pragma unroll 4
    for (int c = 0; c < NHID / 4; ++c) {
        v4f hv = hp[c], av = as[c], dv = ad[c];
        sa += hv.x * av.x + hv.y * av.y + hv.z * av.z + hv.w * av.w;
        sd += hv.x * dv.x + hv.y * dv.y + hv.z * dv.z + hv.w * dv.w;
    }
    a_s[t] = sa; a_d[t] = sd;
}

// segment max of leaky-relu'd edge scores (monotone-uint atomicMax)
__global__ void k_edge_max(const int* __restrict__ src, const int* __restrict__ dst,
                           const float* __restrict__ a_s, const float* __restrict__ a_d,
                           unsigned int* mbits, int E, int Etot) {
    int t = blockIdx.x * blockDim.x + threadIdx.x;
    if (t >= Etot * HEADS) return;
    int e = t >> 2, h0 = t & 3;
    int s = (e < E) ? src[e] : (e - E);
    int d = (e < E) ? dst[e] : (e - E);
    float v = a_s[s * HEADS + h0] + a_d[d * HEADS + h0];
    v = (v > 0.f) ? v : NEG_SLOPE * v;
    atomicMax(&mbits[d * HEADS + h0], fkey(v));
}

// exp(e - max) and segment sum of denominators
__global__ void k_edge_exp(const int* __restrict__ src, const int* __restrict__ dst,
                           const float* __restrict__ a_s, const float* __restrict__ a_d,
                           const unsigned int* __restrict__ mbits,
                           float* exbuf, float* den, int E, int Etot) {
    int t = blockIdx.x * blockDim.x + threadIdx.x;
    if (t >= Etot * HEADS) return;
    int e = t >> 2, h0 = t & 3;
    int s = (e < E) ? src[e] : (e - E);
    int d = (e < E) ? dst[e] : (e - E);
    float v = a_s[s * HEADS + h0] + a_d[d * HEADS + h0];
    v = (v > 0.f) ? v : NEG_SLOPE * v;
    float m = funkey(mbits[d * HEADS + h0]);
    float ex = __expf(v - m);
    exbuf[t] = ex;
    atomicAdd(&den[d * HEADS + h0], ex);
}

// alpha = ex / den[dst]
__global__ void k_norm(const int* __restrict__ dst, float* exbuf,
                       const float* __restrict__ den, int E, int Etot) {
    int t = blockIdx.x * blockDim.x + threadIdx.x;
    if (t >= Etot * HEADS) return;
    int e = t >> 2, h0 = t & 3;
    int d = (e < E) ? dst[e] : (e - E);
    exbuf[t] = exbuf[t] / den[d * HEADS + h0];
}

// message scatter: out[dst] += alpha * h[src]
// block = one edge; thread t -> head t/32, channels 4*(t%32)..+3 (b128 gather)
__global__ void __launch_bounds__(128)
k_scatter(const int* __restrict__ src, const int* __restrict__ dst,
          const float* __restrict__ hbuf, const float* __restrict__ alpha,
          float* outbuf, int E, int Etot) {
    int e = blockIdx.x;
    int t = threadIdx.x;
    int h0 = t >> 5;
    int c4 = (t & 31) << 2;
    int s = (e < E) ? src[e] : (e - E);
    int d = (e < E) ? dst[e] : (e - E);
    float a = alpha[e * HEADS + h0];
    v4f hv = *reinterpret_cast<const v4f*>(hbuf + (size_t)s * HC + h0 * NHID + c4);
    float* op = outbuf + (size_t)d * HC + h0 * NHID + c4;
    atomicAdd(op + 0, a * hv.x);
    atomicAdd(op + 1, a * hv.y);
    atomicAdd(op + 2, a * hv.z);
    atomicAdd(op + 3, a * hv.w);
}

// ELU + group-of-4 mean + GraphCON update; refresh Xbf for the next layer GEMM
__global__ void k_update(const float* __restrict__ outbuf, const float* __restrict__ bias,
                         float* Xcur, float* Ycur, unsigned short* Xbf,
                         float* Xall, float* Yall, int N, int l) {
    int t = blockIdx.x * blockDim.x + threadIdx.x;
    if (t >= N * NHID) return;
    int n = t >> 7, i = t & 127;
    v4f ov = *reinterpret_cast<const v4f*>(outbuf + (size_t)n * HC + i * 4);
    v4f bv = *reinterpret_cast<const v4f*>(bias + i * 4);
    ov += bv;
    float sum = 0.f;
#pragma unroll
    for (int j = 0; j < 4; ++j) {
        float v = ov[j];
        v = (v > 0.f) ? v : (__expf(v) - 1.f);   // ELU
        sum += v;
    }
    float agg = sum * 0.25f;
    float X = Xcur[t], Y = Ycur[t];
    float Y2 = Y + 1.0f * (agg - 1.0f * Y - 1.0f * X);  // DT=ALPHA=GAMMA=1
    float X2 = X + 1.0f * Y2;
    Xcur[t] = X2; Ycur[t] = Y2; Xbf[t] = f2bf(X2);
    Xall[(size_t)n * (NLAYERS + 1) * NHID + (size_t)(l + 1) * NHID + i] = X2;
    Yall[(size_t)n * (NLAYERS + 1) * NHID + (size_t)(l + 1) * NHID + i] = Y2;
}

// logits = Xf @ Wr^T + br (vectorized dot)
__global__ void k_cls(const float* __restrict__ Xcur, const float* __restrict__ Wr,
                      const float* __restrict__ br, float* out, int N) {
    int t = blockIdx.x * blockDim.x + threadIdx.x;
    if (t >= N * NCLASS) return;
    int n = t / NCLASS, k = t % NCLASS;
    const v4f* xp = reinterpret_cast<const v4f*>(Xcur + (size_t)n * NHID);
    const v4f* wp = reinterpret_cast<const v4f*>(Wr + (size_t)k * NHID);
    float s = 0.f;
#pragma unroll 4
    for (int c = 0; c < NHID / 4; ++c) {
        v4f xv = xp[c], wv = wp[c];
        s += xv.x * wv.x + xv.y * wv.y + xv.z * wv.z + xv.w * wv.w;
    }
    out[t] = s + br[k];
}

// ---------------- launch ----------------
extern "C" void kernel_launch(void* const* d_in, const int* in_sizes, int n_in,
                              void* d_out, int out_size, void* d_ws, size_t ws_size,
                              hipStream_t stream) {
    const float* x       = (const float*)d_in[0];
    const int*   src     = (const int*)  d_in[1];
    const int*   dst     = (const int*)  d_in[2];
    const float* W       = (const float*)d_in[3];
    const float* att_src = (const float*)d_in[4];
    const float* att_dst = (const float*)d_in[5];
    const float* bias    = (const float*)d_in[6];
    const float* Wr      = (const float*)d_in[7];
    const float* br      = (const float*)d_in[8];

    const int N    = in_sizes[0] / NHID;
    const int E    = in_sizes[1];
    const int Etot = E + N;

    char* ws = (char*)d_ws;
    size_t off = 0;
    auto alloc = [&](size_t bytes) -> void* {
        void* p = ws + off;
        off = (off + bytes + 255) & ~(size_t)255;
        return p;
    };
    unsigned short* Xbf    = (unsigned short*)alloc((size_t)N * NHID * 2);
    unsigned short* Bpack  = (unsigned short*)alloc((size_t)32 * 4 * 32 * 16 * 2);
    float*          hbuf   = (float*)alloc((size_t)N * HC * 4);
    float*          outbuf = (float*)alloc((size_t)N * HC * 4);
    float*          a_s    = (float*)alloc((size_t)N * HEADS * 4);
    float*          a_d    = (float*)alloc((size_t)N * HEADS * 4);
    unsigned int*   mbits  = (unsigned int*)alloc((size_t)N * HEADS * 4);
    float*          den    = (float*)alloc((size_t)N * HEADS * 4);
    float*          exbuf  = (float*)alloc((size_t)Etot * HEADS * 4);
    float*          Xcur   = (float*)alloc((size_t)N * NHID * 4);
    float*          Ycur   = (float*)alloc((size_t)N * NHID * 4);

    float* outLogits = (float*)d_out;
    float* Xall = outLogits + (size_t)N * NCLASS;
    float* Yall = Xall + (size_t)N * (NLAYERS + 1) * NHID;

    const dim3 b256(256);
    k_init <<<(N * NHID + 255) / 256, b256, 0, stream>>>(x, Xcur, Ycur, Xbf, Xall, Yall, N);
    k_prepB<<<(32 * 4 * 32 * 16 + 255) / 256, b256, 0, stream>>>(W, Bpack);

    for (int l = 0; l < NLAYERS; ++l) {
        k_zero<<<(N * HC / 4 + 255) / 256, b256, 0, stream>>>(outbuf, mbits, den, N);
        dim3 ggrid((N + 15) / 16, HC / 128);
        k_gemm<<<ggrid, dim3(32), 0, stream>>>(Xbf, Bpack, hbuf, N);
        k_att <<<(N * HEADS + 255) / 256, b256, 0, stream>>>(hbuf, att_src, att_dst, a_s, a_d, N);
        int et = (Etot * HEADS + 255) / 256;
        k_edge_max<<<et, b256, 0, stream>>>(src, dst, a_s, a_d, mbits, E, Etot);
        k_edge_exp<<<et, b256, 0, stream>>>(src, dst, a_s, a_d, mbits, exbuf, den, E, Etot);
        k_norm    <<<et, b256, 0, stream>>>(dst, exbuf, den, E, Etot);
        k_scatter <<<Etot, dim3(128), 0, stream>>>(src, dst, hbuf, exbuf, outbuf, E, Etot);
        k_update  <<<(N * NHID + 255) / 256, b256, 0, stream>>>(outbuf, bias, Xcur, Ycur, Xbf, Xall, Yall, N, l);
    }
    k_cls<<<(N * NCLASS + 255) / 256, b256, 0, stream>>>(Xcur, Wr, br, outLogits, N);
}
